// IORTransformer_53790170415764
// MI455X (gfx1250) — matrix operand order repaired and checked by reading, not compiled, not run on hardware
//
#include <hip/hip_runtime.h>
#include <hip/hip_bf16.h>
#include <math.h>

#define DIM   256
#define FFN   512
#define NH    8
#define HD    32
#define LSEQ  16384
#define BATCH 16
#define TL    128
#define NTILE (LSEQ / TL)     // 128 tiles along L
#define EPSF  1e-5f
#define PANEL 8192            // one 32x256 bf16 weight k-panel (elements)
#define NPANEL 8              // 256 / 32

typedef __attribute__((ext_vector_type(16))) __bf16 v16bf;
typedef __attribute__((ext_vector_type(8)))  float  v8f;
typedef __attribute__((ext_vector_type(4)))  unsigned int v4u;
typedef __attribute__((ext_vector_type(8)))  int v8i;
typedef __attribute__((ext_vector_type(4)))  int v4i;

#if defined(__has_builtin)
#if __has_builtin(__builtin_amdgcn_tensor_load_to_lds)
#define HAVE_TDM 1
#endif
#endif

union Frag { v16bf v; unsigned short u[16]; };

__device__ __forceinline__ unsigned short to_bf16(float f) {
  unsigned u = __float_as_uint(f);
  u += 0x7FFFu + ((u >> 16) & 1u);   // round-to-nearest-even
  return (unsigned short)(u >> 16);
}

// A fragment (16x32 bf16, M x K), documented gfx1250 layout:
// lanes 0-15: M=lane, VGPR i holds K = 2i(+8 if i>=4) ; lanes 16-31: +8
// -> two contiguous 8-ushort runs per lane => 2x ds_load_b128
__device__ __forceinline__ void loadA(Frag& a, const unsigned short* sX,
                                      int row0, int lm, int lh, int k0) {
  const unsigned short* rp = sX + (row0 + lm) * DIM + k0 + lh * 8;
#pragma unroll
  for (int i = 0; i < 8; ++i) {
    int ko = 2 * i + ((i >= 4) ? 8 : 0);
    a.u[2 * i]     = rp[ko];
    a.u[2 * i + 1] = rp[ko + 1];
  }
}

#ifdef HAVE_TDM
// Issue a TDM copy of one contiguous 16KB bf16 weight panel into LDS.
// D# group0: count=1, lds_addr, 57-bit global addr, type=2 ("image").
// D# group1: data_size=4B, tensor_dim0=tile_dim0=4096 dwords, 1 row.
__device__ __forceinline__ void tdm_issue_panel(unsigned lds_off,
                                                const unsigned short* g) {
  unsigned long long ga = (unsigned long long)(size_t)g;
  v4u g0;
  g0[0] = 1u;                                           // count=1 (valid)
  g0[1] = lds_off;                                      // lds_addr (bytes)
  g0[2] = (unsigned)(ga & 0xffffffffull);               // global_addr[31:0]
  g0[3] = (unsigned)((ga >> 32) & 0x01ffffffull)        // global_addr[56:32]
        | 0x80000000u;                                  // type=2
  v8i g1;
  g1[0] = (int)(2u << 16);        // workgroup_mask=0, data_size=2 (4 bytes)
  g1[1] = (int)(4096u << 16);     // tensor_dim0[15:0]=4096 (bits 63:48)
  g1[2] = (int)(1u << 16);        // tensor_dim0 hi=0 | tensor_dim1[15:0]=1
  g1[3] = (int)(4096u << 16);     // tensor_dim1 hi=0 | tile_dim0=4096
  g1[4] = 1;                      // tile_dim1=1, tile_dim2=0
  g1[5] = 4096;                   // tensor_dim0_stride lo32
  g1[6] = 0;
  g1[7] = 0;
  v4i z4 = {0, 0, 0, 0};
#if __clang_major__ >= 23
  v8i z8 = {0, 0, 0, 0, 0, 0, 0, 0};
  __builtin_amdgcn_tensor_load_to_lds(g0, g1, z4, z4, z8, 0);
#else
  __builtin_amdgcn_tensor_load_to_lds(g0, g1, z4, z4, 0);
#endif
}
#endif

// One full GEMM pass: acc[16] (16 rows x 256 cols per wave) over K=256,
// weight panels streamed LDS-resident via TDM double-buffering.
__device__ __forceinline__ void gemm_pass(v8f acc[16], const unsigned short* sX,
                                          unsigned short* sB,
                                          const unsigned short* __restrict__ gw,
                                          int row0, int lm, int lh, int tid)
{
#ifdef HAVE_TDM
  const unsigned lds0 = (unsigned)(size_t)(void*)sB;
  for (int p = 0; p < NPANEL; ++p) {
    if (tid == 0) {
      if (p == 0) tdm_issue_panel(lds0, gw);            // prime buffer 0
      if (p + 1 < NPANEL) {
        tdm_issue_panel(lds0 + ((p + 1) & 1) * (PANEL * 2),
                        gw + (size_t)(p + 1) * PANEL);  // prefetch next panel
        __builtin_amdgcn_s_wait_tensorcnt(1);           // oldest (p) done
      } else {
        __builtin_amdgcn_s_wait_tensorcnt(0);
      }
    }
    __syncthreads();                                    // panel p visible to all
    const unsigned short* bpan = sB + (p & 1) * PANEL;
    Frag a; loadA(a, sX, row0, lm, lh, 32 * p);
#pragma unroll
    for (int t = 0; t < 16; ++t) {
      Frag bf;
      const unsigned short* bp = bpan + (16 * t + lm) * 32 + lh * 8;
#pragma unroll
      for (int j = 0; j < 16; ++j) bf.u[j] = bp[j + ((j >= 8) ? 8 : 0)];
      acc[t] = __builtin_amdgcn_wmma_f32_16x16x32_bf16(
          false, a.v, false, bf.v, (short)0, acc[t], false, false);
    }
    __syncthreads();              // done reading buf(p&1) before TDM reuses it
  }
#else
  for (int p = 0; p < NPANEL; ++p) {
    __syncthreads();
    const unsigned long long* src = (const unsigned long long*)(gw + (size_t)p * PANEL);
    unsigned long long* dst = (unsigned long long*)sB;
    for (int e = tid; e < PANEL / 4; e += 256) dst[e] = src[e];
    __syncthreads();
    Frag a; loadA(a, sX, row0, lm, lh, 32 * p);
#pragma unroll
    for (int t = 0; t < 16; ++t) {
      Frag bf;
      const unsigned short* bp = sB + (16 * t + lm) * 32 + lh * 8;
#pragma unroll
      for (int j = 0; j < 16; ++j) bf.u[j] = bp[j + ((j >= 8) ? 8 : 0)];
      acc[t] = __builtin_amdgcn_wmma_f32_16x16x32_bf16(
          false, a.v, false, bf.v, (short)0, acc[t], false, false);
    }
  }
  __syncthreads();
#endif
}

// ---------------------------------------------------------------------------
// Kernel 0: pack W_ip / W_kv(K) / W_kv(V) to bf16, pre-transposed panels:
// wpack[seg][panel p][n][k] = bf16(W[32p+k][n])   (seg: 0=ip, 1=K, 2=V)
// ---------------------------------------------------------------------------
__global__ __launch_bounds__(256)
void packw_kernel(const float* __restrict__ W_ip, const float* __restrict__ W_kv,
                  unsigned short* __restrict__ wpack)
{
  const int e = blockIdx.x * 256 + threadIdx.x;       // < 3*131072
  const int seg = e >> 17;
  const int r = e & 131071;
  const int p = r >> 13;
  const int q = r & 8191;
  const int n = q >> 5, k = q & 31;
  float v;
  if (seg == 0)      v = W_ip[(size_t)(32 * p + k) * DIM + n];
  else if (seg == 1) v = W_kv[(size_t)(32 * p + k) * (2 * DIM) + n];
  else               v = W_kv[(size_t)(32 * p + k) * (2 * DIM) + DIM + n];
  wpack[e] = to_bf16(v);
}

// ---------------------------------------------------------------------------
// Kernel 1: q = ln(query, g_q, be_q) @ W_q + b_q      (B=16 rows)
// ---------------------------------------------------------------------------
__global__ __launch_bounds__(256)
void qproj_kernel(const float* __restrict__ query, const float* __restrict__ g_q,
                  const float* __restrict__ be_q, const float* __restrict__ W_q,
                  const float* __restrict__ b_q, float* __restrict__ qv)
{
  __shared__ float sx[DIM];
  __shared__ float red[8];
  const int b = blockIdx.x, tid = threadIdx.x;
  const int lane = tid & 31, wv = tid >> 5;
  float x = query[b * DIM + tid];

  float v = x;
#pragma unroll
  for (int m = 16; m >= 1; m >>= 1) v += __shfl_xor(v, m, 32);
  if (lane == 0) red[wv] = v;
  __syncthreads();
  float tot = 0.f;
#pragma unroll
  for (int j = 0; j < 8; ++j) tot += red[j];
  const float mean = tot * (1.f / DIM);
  __syncthreads();

  const float d = x - mean;
  float v2 = d * d;
#pragma unroll
  for (int m = 16; m >= 1; m >>= 1) v2 += __shfl_xor(v2, m, 32);
  if (lane == 0) red[wv] = v2;
  __syncthreads();
  float tv = 0.f;
#pragma unroll
  for (int j = 0; j < 8; ++j) tv += red[j];
  const float inv = rsqrtf(tv * (1.f / DIM) + EPSF);

  sx[tid] = d * inv * g_q[tid] + be_q[tid];
  __syncthreads();

  float o = b_q[tid];
  for (int c = 0; c < DIM; ++c) o += sx[c] * W_q[c * DIM + tid];
  qv[b * DIM + tid] = o;
}

// ---------------------------------------------------------------------------
// Kernel 2: fused  relu(mem@W_ip+b)->LN -> @W_kv -> K,V -> online-softmax
// partials per (b, head, L-tile).   One WG = one (b, 128-row tile), 8 waves.
// ---------------------------------------------------------------------------
__global__ __launch_bounds__(256)
void fused_kernel(const float* __restrict__ mem,  const float* __restrict__ ior,
                  const unsigned short* __restrict__ wpack,
                  const float* __restrict__ b_ip,
                  const float* __restrict__ g_ip, const float* __restrict__ be_ip,
                  const float* __restrict__ b_kv,
                  const float* __restrict__ qv,
                  float* __restrict__ partMax, float* __restrict__ partSum,
                  float* __restrict__ partV)
{
  extern __shared__ unsigned char smem[];
  unsigned short* sX   = (unsigned short*)smem;        // 128x256 bf16 (A -> X)
  unsigned short* sB   = sX + TL * DIM;                // 2 x 16KB panel buffers
  float*          sQ   = (float*)(sB + 2 * PANEL);     // 256 q row
  float*          sS   = sQ + DIM;                     // 128x8 scores->weights
  float*          sVac = sS + TL * NH;                 // 8x32 head-V accum

  const int tid  = threadIdx.x;
  const int wv   = tid >> 5;
  const int lane = tid & 31;
  const int lm   = lane & 15;
  const int lh   = lane >> 4;
  const int b    = blockIdx.x / NTILE;
  const int tile = blockIdx.x % NTILE;
  const int l0   = tile * TL;
  const int row0 = 16 * wv;
  const float scale = 0.17677669529663687f;   // 1/sqrt(32)

  const unsigned short* wp1 = wpack;                 // W_ip panels
  const unsigned short* wpK = wpack + 131072;        // W_kv K-half panels
  const unsigned short* wpV = wpack + 262144;        // W_kv V-half panels

  if (tid < DIM) sQ[tid] = qv[b * DIM + tid];
  {
    const float* mp = mem + ((size_t)b * LSEQ + l0) * DIM;
    for (int e = tid; e < TL * DIM; e += 256) sX[e] = to_bf16(mp[e]);
  }
  __syncthreads();

  // ---------------- GEMM1: Y = mem_tile @ W_ip ----------------
  v8f acc[16];
#pragma unroll
  for (int t = 0; t < 16; ++t)
#pragma unroll
    for (int e = 0; e < 8; ++e) acc[t][e] = 0.f;
  gemm_pass(acc, sX, sB, wp1, row0, lm, lh, tid);

  // ---------------- bias + ReLU + LayerNorm (C-layout) ----------------
  {
    float s1[8], s2[8];
#pragma unroll
    for (int i = 0; i < 8; ++i) { s1[i] = 0.f; s2[i] = 0.f; }
#pragma unroll
    for (int t = 0; t < 16; ++t) {
      const int n = 16 * t + lm;
      const float bi = b_ip[n];
#pragma unroll
      for (int i = 0; i < 8; ++i) {
        float y = fmaxf(acc[t][i] + bi, 0.f);
        acc[t][i] = y;
        s1[i] += y; s2[i] += y * y;
      }
    }
#pragma unroll
    for (int i = 0; i < 8; ++i)
#pragma unroll
      for (int m = 8; m >= 1; m >>= 1) {   // reduce across row-mates (half-wave)
        s1[i] += __shfl_xor(s1[i], m, 32);
        s2[i] += __shfl_xor(s2[i], m, 32);
      }
#pragma unroll
    for (int t = 0; t < 16; ++t) {
      const int n = 16 * t + lm;
      const float g = g_ip[n], be = be_ip[n];
#pragma unroll
      for (int i = 0; i < 8; ++i) {
        const float mean = s1[i] * (1.f / DIM);
        const float var  = s2[i] * (1.f / DIM) - mean * mean;
        const float inv  = rsqrtf(var + EPSF);
        const float xn   = (acc[t][i] - mean) * inv * g + be;
        const int r = row0 + i + 8 * lh;     // wave-private rows: no barrier
        sX[r * DIM + n] = to_bf16(xn);
      }
    }
  }

  // ---------------- GEMM2 (K half) ----------------
  v8f kac[16];
#pragma unroll
  for (int t = 0; t < 16; ++t)
#pragma unroll
    for (int e = 0; e < 8; ++e) kac[t][e] = 0.f;
  gemm_pass(kac, sX, sB, wpK, row0, lm, lh, tid);
#pragma unroll
  for (int t = 0; t < 16; ++t) {
    const float bk = b_kv[16 * t + lm];
#pragma unroll
    for (int i = 0; i < 8; ++i) kac[t][i] += bk;
  }

  // scores: s[r][h] = scale * ior[b,h,l0+r] * sum_d q[32h+d]*K[r][32h+d]
#pragma unroll
  for (int h = 0; h < NH; ++h) {
    const float qa = sQ[32 * h + lm];
    const float qb = sQ[32 * h + 16 + lm];
#pragma unroll
    for (int i = 0; i < 8; ++i) {
      float p = kac[2 * h][i] * qa + kac[2 * h + 1][i] * qb;
#pragma unroll
      for (int m = 8; m >= 1; m >>= 1) p += __shfl_xor(p, m, 32);
      if (lm == h) {
        const int r = row0 + i + 8 * lh;
        const float io = ior[((size_t)(b * NH + h)) * LSEQ + l0 + r];
        sS[r * NH + h] = p * scale * io;
      }
    }
  }

  // ---------------- GEMM2 (V half) ----------------
  v8f vac[16];
#pragma unroll
  for (int t = 0; t < 16; ++t)
#pragma unroll
    for (int e = 0; e < 8; ++e) vac[t][e] = 0.f;
  gemm_pass(vac, sX, sB, wpV, row0, lm, lh, tid);
#pragma unroll
  for (int t = 0; t < 16; ++t) {
    const float bv = b_kv[DIM + 16 * t + lm];
#pragma unroll
    for (int i = 0; i < 8; ++i) vac[t][i] += bv;
  }
  __syncthreads();   // all sS writes visible

  // ---------------- per-head tile softmax partials (wave == head) ---------
  {
    const int h = wv;
    float sc[4];
    float mx = -3.4e38f;
#pragma unroll
    for (int j = 0; j < 4; ++j) {
      sc[j] = sS[(lane + 32 * j) * NH + h];
      mx = fmaxf(mx, sc[j]);
    }
#pragma unroll
    for (int m = 16; m >= 1; m >>= 1) mx = fmaxf(mx, __shfl_xor(mx, m, 32));
    float sum = 0.f;
#pragma unroll
    for (int j = 0; j < 4; ++j) {
      const float e = __expf(sc[j] - mx);
      sS[(lane + 32 * j) * NH + h] = e;     // store exp weight back
      sum += e;
    }
#pragma unroll
    for (int m = 16; m >= 1; m >>= 1) sum += __shfl_xor(sum, m, 32);
    if (lane == 0) {
      partMax[((size_t)(b * NH + h)) * NTILE + tile] = mx;
      partSum[((size_t)(b * NH + h)) * NTILE + tile] = sum;
    }
  }
  __syncthreads();
  sVac[tid] = 0.f;      // 8 heads x 32 dims
  __syncthreads();

  // ---------------- V accumulation straight from WMMA accumulators --------
#pragma unroll
  for (int t = 0; t < 16; ++t) {
    const int h = t >> 1;
    float p = 0.f;
#pragma unroll
    for (int i = 0; i < 8; ++i) {
      const int r = row0 + i + 8 * lh;
      p += sS[r * NH + h] * vac[t][i];
    }
    atomicAdd(&sVac[h * HD + 16 * (t & 1) + lm], p);   // ds_add_f32
  }
  __syncthreads();

  {
    const int h = tid >> 5, d = tid & 31;
    partV[(((size_t)(b * NH + h)) * NTILE + tile) * HD + d] = sVac[tid];
  }
}

// ---------------------------------------------------------------------------
// Kernel 3: reduce tile partials -> attention out; residual + LN + GELU FFN
// ---------------------------------------------------------------------------
__global__ __launch_bounds__(256)
void finalize_kernel(const float* __restrict__ query,
                     const float* __restrict__ g_f, const float* __restrict__ be_f,
                     const float* __restrict__ W1, const float* __restrict__ b1,
                     const float* __restrict__ W2, const float* __restrict__ b2,
                     const float* __restrict__ partMax, const float* __restrict__ partSum,
                     const float* __restrict__ partV, float* __restrict__ out)
{
  __shared__ float sO[DIM];
  __shared__ float sXv[DIM];
  __shared__ float sH[DIM];
  __shared__ float sF[FFN];
  __shared__ float red[8];
  const int b = blockIdx.x, tid = threadIdx.x;
  const int wv = tid >> 5, lane = tid & 31;

  {
    const int h = wv;
    const float* pm = partMax + ((size_t)(b * NH + h)) * NTILE;
    const float* ps = partSum + ((size_t)(b * NH + h)) * NTILE;
    const float* pv = partV + (((size_t)(b * NH + h)) * NTILE) * HD;
    float mx = -3.4e38f;
    for (int t = lane; t < NTILE; t += 32) mx = fmaxf(mx, pm[t]);
#pragma unroll
    for (int m = 16; m >= 1; m >>= 1) mx = fmaxf(mx, __shfl_xor(mx, m, 32));
    float S = 0.f;
    for (int t = lane; t < NTILE; t += 32) S += ps[t] * __expf(pm[t] - mx);
#pragma unroll
    for (int m = 16; m >= 1; m >>= 1) S += __shfl_xor(S, m, 32);
    float o = 0.f;
    for (int t = 0; t < NTILE; ++t) o += __expf(pm[t] - mx) * pv[t * HD + lane];
    sO[h * HD + lane] = o / S;
  }
  __syncthreads();

  const float x = sO[tid] + query[b * DIM + tid];
  sXv[tid] = x;

  float v = x;
#pragma unroll
  for (int m = 16; m >= 1; m >>= 1) v += __shfl_xor(v, m, 32);
  if (lane == 0) red[wv] = v;
  __syncthreads();
  float tot = 0.f;
#pragma unroll
  for (int j = 0; j < 8; ++j) tot += red[j];
  const float mean = tot * (1.f / DIM);
  __syncthreads();
  const float dd = x - mean;
  float v2 = dd * dd;
#pragma unroll
  for (int m = 16; m >= 1; m >>= 1) v2 += __shfl_xor(v2, m, 32);
  if (lane == 0) red[wv] = v2;
  __syncthreads();
  float tv = 0.f;
#pragma unroll
  for (int j = 0; j < 8; ++j) tv += red[j];
  const float inv = rsqrtf(tv * (1.f / DIM) + EPSF);
  sH[tid] = dd * inv * g_f[tid] + be_f[tid];
  __syncthreads();

#pragma unroll
  for (int rep = 0; rep < 2; ++rep) {
    const int j = tid + rep * 256;
    float a = b1[j];
    for (int c = 0; c < DIM; ++c) a += sH[c] * W1[c * FFN + j];
    sF[j] = a * 0.5f * (1.f + erff(a * 0.70710678118f));   // exact GELU
  }
  __syncthreads();

  float o2 = b2[tid];
  for (int j = 0; j < FFN; ++j) o2 += sF[j] * W2[j * DIM + tid];
  out[b * DIM + tid] = sXv[tid] + o2;
}

// ---------------------------------------------------------------------------
extern "C" void kernel_launch(void* const* d_in, const int* in_sizes, int n_in,
                              void* d_out, int out_size, void* d_ws, size_t ws_size,
                              hipStream_t stream) {
  (void)in_sizes; (void)n_in; (void)out_size; (void)ws_size;
  const float* query = (const float*)d_in[0];
  const float* mem   = (const float*)d_in[1];
  const float* ior   = (const float*)d_in[2];
  const float* W_ip  = (const float*)d_in[3];
  const float* b_ip  = (const float*)d_in[4];
  const float* g_ip  = (const float*)d_in[5];
  const float* be_ip = (const float*)d_in[6];
  const float* W_q   = (const float*)d_in[7];
  const float* b_q   = (const float*)d_in[8];
  const float* W_kv  = (const float*)d_in[9];
  const float* b_kv  = (const float*)d_in[10];
  const float* g_q   = (const float*)d_in[11];
  const float* be_q  = (const float*)d_in[12];
  const float* g_f   = (const float*)d_in[13];
  const float* be_f  = (const float*)d_in[14];
  const float* W1    = (const float*)d_in[15];
  const float* b1    = (const float*)d_in[16];
  const float* W2    = (const float*)d_in[17];
  const float* b2    = (const float*)d_in[18];
  float* out = (float*)d_out;

  float* ws      = (float*)d_ws;
  float* qv      = ws;                               // 16*256
  float* partMax = qv + BATCH * DIM;                 // 16*8*128
  float* partSum = partMax + BATCH * NH * NTILE;
  float* partV   = partSum + BATCH * NH * NTILE;     // 16*8*128*32
  unsigned short* wpack = (unsigned short*)(partV + (size_t)BATCH * NH * NTILE * HD);

  packw_kernel<<<1536, 256, 0, stream>>>(W_ip, W_kv, wpack);
  qproj_kernel<<<BATCH, 256, 0, stream>>>(query, g_q, be_q, W_q, b_q, qv);

  const size_t smem = (size_t)TL * DIM * 2      // sX      64 KB
                    + (size_t)2 * PANEL * 2     // sB      32 KB (double buffer)
                    + (size_t)DIM * 4           // sQ       1 KB
                    + (size_t)TL * NH * 4       // sS       4 KB
                    + (size_t)256 * 4;          // sVac     1 KB  => 104448 B
  fused_kernel<<<BATCH * NTILE, 256, smem, stream>>>(
      mem, ior, wpack, b_ip, g_ip, be_ip, b_kv, qv,
      partMax, partSum, partV);

  finalize_kernel<<<BATCH, 256, 0, stream>>>(
      query, g_f, be_f, W1, b1, W2, b2, partMax, partSum, partV, out);
}
